// AugmentedNeuralODE_75058848465048
// MI455X (gfx1250) — matrix operand order repaired and checked by reading, not compiled, not run on hardware
//
#include <hip/hip_runtime.h>
#include <hip/hip_bf16.h>
#include <math.h>

// CDNA5 WMMA types
typedef __attribute__((ext_vector_type(16))) _Float16 v16h;
typedef __attribute__((ext_vector_type(8)))  _Float16 v8h;
typedef __attribute__((ext_vector_type(8)))  float    v8f;

#define WAVES 4            // waves per block (each wave owns 16 batch rows)
#define HSTR  136          // LDS activation row stride in halves (bank spread)

__device__ __forceinline__ v8f wmma_f16(v16h a, v16h b, v8f c) {
  // (neg_a, A, neg_b, B, c_mod, C, reuse_a, reuse_b)
  return __builtin_amdgcn_wmma_f32_16x16x32_f16(false, a, false, b, (short)0, c,
                                                false, false);
}

// Branch-free tanh: native v_tanh_f32 on gfx1250 (probe-confirmed last round).
__device__ __forceinline__ float fast_tanh(float x) {
#if __has_builtin(__builtin_amdgcn_tanhf)
  return __builtin_amdgcn_tanhf(x);               // v_tanh_f32
#else
  float e = __builtin_amdgcn_exp2f(x * 2.8853900817779268f);  // v_exp_f32
  float r = __builtin_amdgcn_rcpf(e + 1.0f);                  // v_rcp_f32
  return 1.0f - 2.0f * r;
#endif
}

__global__ __launch_bounds__(WAVES * 32)
void AugmentedNeuralODE_kernel(const float* __restrict__ r0,
                               const float* __restrict__ tg,
                               const float* __restrict__ W1,
                               const float* __restrict__ b1,
                               const float* __restrict__ W2,
                               const float* __restrict__ b2,
                               const float* __restrict__ W3,
                               const float* __restrict__ b3,
                               float* __restrict__ out)
{
  // Weights pre-swizzled into WMMA B-fragment layout:
  //   fragment f, lane l, half i  ->  W[K][N],  N = tileN*16 + (l&15),
  //   K = tileK*32 + (l>>4)*16 + i   (lanes 0-15: K=0..15, lanes 16-31: K=16..31)
  // b1 is folded into W1 as K=5 (paired with a constant-1 input feature), so the
  // layer-1 accumulator starts at inline 0.
  __shared__ __attribute__((aligned(32))) _Float16 sW1[8 * 32 * 16];        //  8 KB
  __shared__ __attribute__((aligned(32))) _Float16 sW2[8 * 4 * 32 * 16];    // 32 KB
  __shared__ __attribute__((aligned(32))) _Float16 sW3[4 * 32 * 16];        //  4 KB
  __shared__ float sb2[128], sb3[2];
  __shared__ __attribute__((aligned(32))) _Float16 hbuf[WAVES][16 * HSTR];  // activations
  __shared__ float kbuf[WAVES][32];                                         // per-row (kx,ky)

  const int tid  = threadIdx.x;
  const int lane = tid & 31;
  const int wave = tid >> 5;
  const int m    = lane & 15;   // row within tile / column within N-tile
  const int kh   = lane >> 4;   // half-select (0: lanes 0-15, 1: lanes 16-31)

  // ---- cooperative weight packing (once per block) ----
  for (int idx = tid; idx < 8 * 4 * 32; idx += blockDim.x) {   // W2: [nt][kt][lane]
    const int nt = idx >> 7, kt = (idx >> 5) & 3, ln = idx & 31;
    const int n = nt * 16 + (ln & 15);
    const int kbase = kt * 32 + (ln >> 4) * 16;
    #pragma unroll
    for (int i = 0; i < 16; ++i)
      sW2[idx * 16 + i] = (_Float16)W2[(kbase + i) * 128 + n];
  }
  for (int idx = tid; idx < 8 * 32; idx += blockDim.x) {       // W1: K padded 5 -> 32
    const int nt = idx >> 5, ln = idx & 31;
    const int n = nt * 16 + (ln & 15);
    const int kbase = (ln >> 4) * 16;
    #pragma unroll
    for (int i = 0; i < 16; ++i) {
      const int k = kbase + i;
      float w = 0.0f;
      if (k < 5)       w = W1[k * 128 + n];
      else if (k == 5) w = b1[n];                 // bias row (feature 5 == 1.0)
      sW1[idx * 16 + i] = (_Float16)w;
    }
  }
  for (int idx = tid; idx < 4 * 32; idx += blockDim.x) {       // W3: N padded 2 -> 16
    const int kt = idx >> 5, ln = idx & 31;
    const int n = ln & 15;
    const int kbase = kt * 32 + (ln >> 4) * 16;
    #pragma unroll
    for (int i = 0; i < 16; ++i)
      sW3[idx * 16 + i] = (_Float16)((n < 2) ? W3[(kbase + i) * 2 + n] : 0.0f);
  }
  for (int i = tid; i < 128; i += blockDim.x) sb2[i] = b2[i];
  if (tid < 2) sb3[tid] = b3[tid];
  __syncthreads();

  // ---- per-wave state: 16 batch rows; lanes m and m+16 mirror the same row ----
  const int row = (blockIdx.x * WAVES + wave) * 16 + m;
  float xs = r0[row * 2 + 0];
  float ys = r0[row * 2 + 1];
  float tt[8];
  #pragma unroll
  for (int i = 0; i < 8; ++i) tt[i] = tg[i];
  if (lane < 16) { out[row * 16 + 0] = xs; out[row * 16 + 1] = ys; }

  _Float16* hb = &hbuf[wave][0];
  float*    kb = &kbuf[wave][0];
  const int rb = (lane < 16) ? 0 : 8;   // C/D fragment row base for this half-wave

  // one MLP evaluation: (t, x, y) for this wave's 16 rows -> (kx, ky) per row
  auto evalf = [&](float ts, float px, float py, float& kx, float& ky) {
    // ---- layer 1: inp[16,32] (x, y, 0, 0, t, 1, zero-pad) @ [W1; b1] ----
    // A layout (16-bit, 16x32): lane<16 holds K=0..15 -> features live here.
    v16h a1;
    #pragma unroll
    for (int i = 0; i < 16; ++i) a1[i] = (_Float16)0.0f;
    if (lane < 16) {
      a1[0] = (_Float16)px;  a1[1] = (_Float16)py;
      a1[4] = (_Float16)ts;  a1[5] = (_Float16)1.0f;   // bias feature
    }
    #pragma unroll
    for (int nt = 0; nt < 8; ++nt) {
      v16h bf = *(const v16h*)&sW1[(nt * 32 + lane) * 16];
      v8f c = {};                                  // inline-0 SRC2
      c = wmma_f16(a1, bf, c);
      #pragma unroll
      for (int j = 0; j < 8; ++j)
        hb[(rb + j) * HSTR + nt * 16 + m] = (_Float16)fast_tanh(c[j]);
    }

    // ---- layer 2: h[16,128] @ W2[128,128] ----
    v16h a[4];
    #pragma unroll
    for (int kt = 0; kt < 4; ++kt) {   // A frag: halves 0..7 = K base.., 8..15 = +8
      const int base = kt * 32 + kh * 16;
      v8h lo = *(const v8h*)&hb[m * HSTR + base];
      v8h hi = *(const v8h*)&hb[m * HSTR + base + 8];
      #pragma unroll
      for (int j = 0; j < 8; ++j) { a[kt][j] = lo[j]; a[kt][j + 8] = hi[j]; }
    }
    #pragma unroll
    for (int nt = 0; nt < 8; ++nt) {
      const float bias = sb2[nt * 16 + m];
      v8f c;
      #pragma unroll
      for (int j = 0; j < 8; ++j) c[j] = bias;
      #pragma unroll
      for (int kt = 0; kt < 4; ++kt) {
        v16h bf = *(const v16h*)&sW2[((nt * 4 + kt) * 32 + lane) * 16];
        c = wmma_f16(a[kt], bf, c);
      }
      #pragma unroll
      for (int j = 0; j < 8; ++j)
        hb[(rb + j) * HSTR + nt * 16 + m] = (_Float16)fast_tanh(c[j]);
    }

    // ---- layer 3: h2[16,128] @ W3[128,2] (N padded to 16) ----
    #pragma unroll
    for (int kt = 0; kt < 4; ++kt) {
      const int base = kt * 32 + kh * 16;
      v8h lo = *(const v8h*)&hb[m * HSTR + base];
      v8h hi = *(const v8h*)&hb[m * HSTR + base + 8];
      #pragma unroll
      for (int j = 0; j < 8; ++j) { a[kt][j] = lo[j]; a[kt][j + 8] = hi[j]; }
    }
    v8f c3;
    const float bias3 = (m < 2) ? sb3[m] : 0.0f;
    #pragma unroll
    for (int j = 0; j < 8; ++j) c3[j] = bias3;
    #pragma unroll
    for (int kt = 0; kt < 4; ++kt) {
      v16h bf = *(const v16h*)&sW3[(kt * 32 + lane) * 16];
      c3 = wmma_f16(a[kt], bf, c3);
    }
    // scatter velocities (C layout: lane N holds rows rb..rb+7) -> per-row regs
    if (m < 2) {
      #pragma unroll
      for (int j = 0; j < 8; ++j) kb[(rb + j) * 2 + m] = c3[j];
    }
    kx = kb[m * 2 + 0];          // same-wave LDS is in-order: store->load safe
    ky = kb[m * 2 + 1];
  };

  // ---- dopri5, 2 fixed substeps per interval ----
  for (int ti = 0; ti < 7; ++ti) {
    const float t0 = tt[ti];
    const float dt = (tt[ti + 1] - t0) * 0.5f;
    for (int sub = 0; sub < 2; ++sub) {
      const float ta = t0 + dt * (float)sub;
      float k1x, k1y, k2x, k2y, k3x, k3y, k4x, k4y, k5x, k5y, k6x, k6y;
      evalf(ta, xs, ys, k1x, k1y);
      evalf(ta + 0.2f * dt, xs + dt * 0.2f * k1x, ys + dt * 0.2f * k1y, k2x, k2y);
      evalf(ta + 0.3f * dt,
            xs + dt * (0.075f * k1x + 0.225f * k2x),
            ys + dt * (0.075f * k1y + 0.225f * k2y), k3x, k3y);
      evalf(ta + 0.8f * dt,
            xs + dt * ((44.0f/45.0f)*k1x - (56.0f/15.0f)*k2x + (32.0f/9.0f)*k3x),
            ys + dt * ((44.0f/45.0f)*k1y - (56.0f/15.0f)*k2y + (32.0f/9.0f)*k3y),
            k4x, k4y);
      evalf(ta + (8.0f/9.0f) * dt,
            xs + dt * ((19372.0f/6561.0f)*k1x - (25360.0f/2187.0f)*k2x
                     + (64448.0f/6561.0f)*k3x - (212.0f/729.0f)*k4x),
            ys + dt * ((19372.0f/6561.0f)*k1y - (25360.0f/2187.0f)*k2y
                     + (64448.0f/6561.0f)*k3y - (212.0f/729.0f)*k4y),
            k5x, k5y);
      evalf(ta + dt,
            xs + dt * ((9017.0f/3168.0f)*k1x - (355.0f/33.0f)*k2x
                     + (46732.0f/5247.0f)*k3x + (49.0f/176.0f)*k4x
                     - (5103.0f/18656.0f)*k5x),
            ys + dt * ((9017.0f/3168.0f)*k1y - (355.0f/33.0f)*k2y
                     + (46732.0f/5247.0f)*k3y + (49.0f/176.0f)*k4y
                     - (5103.0f/18656.0f)*k5y),
            k6x, k6y);
      xs += dt * ((35.0f/384.0f)*k1x + (500.0f/1113.0f)*k3x + (125.0f/192.0f)*k4x
                - (2187.0f/6784.0f)*k5x + (11.0f/84.0f)*k6x);
      ys += dt * ((35.0f/384.0f)*k1y + (500.0f/1113.0f)*k3y + (125.0f/192.0f)*k4y
                - (2187.0f/6784.0f)*k5y + (11.0f/84.0f)*k6y);
    }
    if (lane < 16) {
      out[row * 16 + (ti + 1) * 2 + 0] = xs;
      out[row * 16 + (ti + 1) * 2 + 1] = ys;
    }
  }
}

extern "C" void kernel_launch(void* const* d_in, const int* in_sizes, int n_in,
                              void* d_out, int out_size, void* d_ws, size_t ws_size,
                              hipStream_t stream) {
  (void)n_in; (void)out_size; (void)d_ws; (void)ws_size;
  const float* r0 = (const float*)d_in[0];
  const float* tg = (const float*)d_in[1];
  const float* W1 = (const float*)d_in[2];
  const float* b1 = (const float*)d_in[3];
  const float* W2 = (const float*)d_in[4];
  const float* b2 = (const float*)d_in[5];
  const float* W3 = (const float*)d_in[6];
  const float* b3 = (const float*)d_in[7];
  float* out = (float*)d_out;

  const int Bn = in_sizes[0] / 2;            // 16384
  const int blocks = Bn / (WAVES * 16);      // 256 blocks x 128 threads
  AugmentedNeuralODE_kernel<<<blocks, WAVES * 32, 0, stream>>>(
      r0, tg, W1, b1, W2, b2, W3, b3, out);
}
